// RNN_13726715478279
// MI455X (gfx1250) — matrix operand order repaired and checked by reading, not compiled, not run on hardware
//
#include <hip/hip_runtime.h>
#include <hip/hip_bf16.h>

// ---------------------------------------------------------------------------
// Two-layer tanh RNN, persistent-cooperative kernel for gfx1250 (MI455X).
// B=64, T=512, H=512, I=128, W_OUT=32.
//
// 16 persistent workgroups x 256 threads (8 waves). Each WG owns a 32-column
// slice of H. Weight slices cached in LDS (104KB/WGP). Hidden state ping-pong
// buffered in global f16; grid barrier (atomics, agent scope) twice per step.
// Matrix math: v_wmma_f32_16x16x32_f16, two accumulators per tile for ILP.
// ---------------------------------------------------------------------------

typedef __attribute__((ext_vector_type(16))) _Float16 v16h;
typedef __attribute__((ext_vector_type(8)))  _Float16 v8h;
typedef __attribute__((ext_vector_type(8)))  float    v8f;

#define B_    64
#define T_    512
#define H_    512
#define I_    128
#define WOUT_ 32
#define NWG   16
#define SLICE 32   // H_ / NWG

// ---- workspace layout (bytes) ---------------------------------------------
constexpr size_t OFF_WIH0 = 0;                                   // [H][I] f16
constexpr size_t OFF_WHH0 = OFF_WIH0 + (size_t)H_ * I_ * 2;      // [H][H] f16
constexpr size_t OFF_WIH1 = OFF_WHH0 + (size_t)H_ * H_ * 2;
constexpr size_t OFF_WHH1 = OFF_WIH1 + (size_t)H_ * H_ * 2;
constexpr size_t OFF_FCW  = OFF_WHH1 + (size_t)H_ * H_ * 2;      // [WOUT][H] f16
constexpr size_t OFF_X    = OFF_FCW + (size_t)WOUT_ * H_ * 2;    // [B][T][I] f16
constexpr size_t OFF_H0   = OFF_X + (size_t)B_ * T_ * I_ * 2;    // [2][B][H] f16
constexpr size_t OFF_H1   = OFF_H0 + (size_t)2 * B_ * H_ * 2;    // [2][B][H] f16
constexpr size_t OFF_BAR  = OFF_H1 + (size_t)2 * B_ * H_ * 2;    // 2x u32 (+pad)
constexpr size_t SMEM_BYTES = (size_t)(SLICE * I_ + 3 * SLICE * H_) * 2; // 106496

// ---- WMMA fragment helpers -------------------------------------------------
// A fragment: 16x32 f16, source row-major with leading dim `ld` (elements).
// Lane l: row m = l&15; VGPR0-3 hold K = kg..kg+7, VGPR4-7 hold K = kg+8..kg+15
// where kg = (l>=16)?16:0  (contiguous K, matching B fragment).
__device__ __forceinline__ v16h load_a_frag(const _Float16* tile, int ld, int lane) {
  const int m  = lane & 15;
  const int kg = (lane >> 4) << 4;           // 0 or 16
  const _Float16* p = tile + (size_t)m * ld + kg;
  v8h lo = *(const v8h*)(p);                 // K = kg .. kg+7
  v8h hi = *(const v8h*)(p + 8);             // K = kg+8 .. kg+15
  return __builtin_shufflevector(lo, hi, 0, 1, 2, 3, 4, 5, 6, 7,
                                         8, 9, 10, 11, 12, 13, 14, 15);
}

// B fragment: 32x16 f16 from a row-major [N][K] weight matrix (W[n][k]).
// Lane l: column n = l&15; 16 contiguous K values starting at (l>=16)?16:0.
__device__ __forceinline__ v16h load_b_frag(const _Float16* tile, int ld, int lane) {
  const int n    = lane & 15;
  const int koff = (lane >> 4) << 4;         // 0 or 16
  return *(const v16h*)(tile + (size_t)n * ld + koff);
}

__device__ __forceinline__ v8f wmma_f16(v16h a, v16h b, v8f c) {
  return __builtin_amdgcn_wmma_f32_16x16x32_f16(
      /*neg_a=*/false, a, /*neg_b=*/false, b,
      /*c_mod=*/(short)0, c, /*reuse_a=*/false, /*reuse_b=*/false);
}

// One 16x16 output tile: acc += A[16,K] * B[K,16], two interleaved
// accumulators to break the WMMA D->C dependency chain.
__device__ __forceinline__ v8f tile_mma(const _Float16* abase, int lda,
                                        const _Float16* bbase, int ldb,
                                        int K, int lane, v8f acc) {
  v8f a0 = {};
  v8f a1 = {};
  for (int kb = 0; kb < K; kb += 64) {
    a0 = wmma_f16(load_a_frag(abase + kb, lda, lane),
                  load_b_frag(bbase + kb, ldb, lane), a0);
    a1 = wmma_f16(load_a_frag(abase + kb + 32, lda, lane),
                  load_b_frag(bbase + kb + 32, ldb, lane), a1);
  }
  return acc + a0 + a1;
}

// ---- device-scope grid barrier (16 resident blocks, sense via generation) --
__device__ __forceinline__ void grid_barrier(unsigned* cnt, unsigned* gen) {
  __threadfence();           // make this block's global stores visible (agent)
  __syncthreads();
  if (threadIdx.x == 0) {
    unsigned g = __hip_atomic_load(gen, __ATOMIC_ACQUIRE, __HIP_MEMORY_SCOPE_AGENT);
    unsigned arrived =
        __hip_atomic_fetch_add(cnt, 1u, __ATOMIC_ACQ_REL, __HIP_MEMORY_SCOPE_AGENT);
    if (arrived == (unsigned)(NWG - 1)) {
      __hip_atomic_store(cnt, 0u, __ATOMIC_RELAXED, __HIP_MEMORY_SCOPE_AGENT);
      __hip_atomic_fetch_add(gen, 1u, __ATOMIC_RELEASE, __HIP_MEMORY_SCOPE_AGENT);
    } else {
      while (__hip_atomic_load(gen, __ATOMIC_ACQUIRE, __HIP_MEMORY_SCOPE_AGENT) == g) {
        __builtin_amdgcn_s_sleep(1);
      }
    }
  }
  __syncthreads();
}

// ---- pre-pass kernels ------------------------------------------------------
__global__ void cvt_f32_to_f16_kernel(const float* __restrict__ src,
                                      _Float16* __restrict__ dst, int n) {
  for (int i = blockIdx.x * blockDim.x + threadIdx.x; i < n;
       i += gridDim.x * blockDim.x)
    dst[i] = (_Float16)src[i];
}

__global__ void zero_u32_kernel(unsigned* __restrict__ dst, int n) {
  for (int i = blockIdx.x * blockDim.x + threadIdx.x; i < n;
       i += gridDim.x * blockDim.x)
    dst[i] = 0u;
}

// ---- main persistent RNN kernel -------------------------------------------
__global__ __launch_bounds__(256, 1) void rnn_persistent(
    const _Float16* __restrict__ hx,     // [B][T][I] f16
    const float*    __restrict__ noise,  // [B][T][2][H] f32
    const _Float16* __restrict__ wih0,   // [H][I]
    const _Float16* __restrict__ whh0,   // [H][H]
    const _Float16* __restrict__ wih1,   // [H][H]
    const _Float16* __restrict__ whh1,   // [H][H]
    const float* __restrict__ b_ih0, const float* __restrict__ b_ih1,
    const float* __restrict__ b_hh0, const float* __restrict__ b_hh1,
    const _Float16* __restrict__ hfc,    // [WOUT][H]
    const float* __restrict__ fc_b,      // [WOUT]
    _Float16* __restrict__ h0buf,        // [2][B][H]
    _Float16* __restrict__ h1buf,        // [2][B][H]
    float* __restrict__ out,             // [B][WOUT]
    unsigned* bar_cnt, unsigned* bar_gen) {
  extern __shared__ char smem_raw[];
  _Float16* sWih0 = (_Float16*)smem_raw;      // [SLICE][I]
  _Float16* sWhh0 = sWih0 + SLICE * I_;       // [SLICE][H]
  _Float16* sWih1 = sWhh0 + SLICE * H_;       // [SLICE][H]
  _Float16* sWhh1 = sWih1 + SLICE * H_;       // [SLICE][H]

  const int wg    = blockIdx.x;       // 0..15 -> hidden column slice
  const int nbase = wg * SLICE;
  const int tid   = threadIdx.x;
  const int wave  = tid >> 5;         // 0..7
  const int lane  = tid & 31;
  const int mbase = (wave & 3) * 16;  // batch-row tile
  const int ncol  = (wave >> 2) * 16; // column tile within slice

  // Per-lane epilogue coordinates (C/D layout: N = lane&15, M = r + 8*(lane>=16))
  const int nrel   = ncol + (lane & 15);
  const int ng     = nbase + nrel;
  const int rowsel = (lane >> 4) << 3;
  const float bias0 = b_ih0[ng] + b_hh0[ng];
  const float bias1 = b_ih1[ng] + b_hh1[ng];

  // ---- one-time: cache this WG's weight slices in LDS (uint4 = 8 f16) ----
  {
    const uint4* g;
    uint4* s;
    g = (const uint4*)(wih0 + (size_t)nbase * I_); s = (uint4*)sWih0;
    for (int i = tid; i < SLICE * I_ / 8; i += 256) s[i] = g[i];
    g = (const uint4*)(whh0 + (size_t)nbase * H_); s = (uint4*)sWhh0;
    for (int i = tid; i < SLICE * H_ / 8; i += 256) s[i] = g[i];
    g = (const uint4*)(wih1 + (size_t)nbase * H_); s = (uint4*)sWih1;
    for (int i = tid; i < SLICE * H_ / 8; i += 256) s[i] = g[i];
    g = (const uint4*)(whh1 + (size_t)nbase * H_); s = (uint4*)sWhh1;
    for (int i = tid; i < SLICE * H_ / 8; i += 256) s[i] = g[i];
  }
  __syncthreads();

  // ---- sequential time loop ----
  for (int t = 0; t < T_; ++t) {
    const _Float16* h0c = h0buf + (size_t)(t & 1) * (B_ * H_);
    _Float16*       h0n = h0buf + (size_t)((t + 1) & 1) * (B_ * H_);
    const _Float16* h1c = h1buf + (size_t)(t & 1) * (B_ * H_);
    _Float16*       h1n = h1buf + (size_t)((t + 1) & 1) * (B_ * H_);

    // warm next step's noise lines into cache
    if (t + 1 < T_) {
      __builtin_prefetch(
          &noise[(((size_t)(mbase + rowsel) * T_ + (t + 1)) * 2) * H_ + ng], 0, 1);
    }

    // ===== layer 0: h0 = tanh(x_t W_ih0^T + h0 W_hh0^T + biases + n0) =====
    v8f acc = {};
    acc = tile_mma(hx + (size_t)mbase * (T_ * I_) + (size_t)t * I_, T_ * I_,
                   sWih0 + (size_t)ncol * I_, I_, I_, lane, acc);
    acc = tile_mma(h0c + (size_t)mbase * H_, H_,
                   sWhh0 + (size_t)ncol * H_, H_, H_, lane, acc);
    for (int r = 0; r < 8; ++r) {
      const int bb = mbase + rowsel + r;
      const float v =
          acc[r] + bias0 + noise[(((size_t)bb * T_ + t) * 2 + 0) * H_ + ng];
      h0n[(size_t)bb * H_ + ng] = (_Float16)tanhf(v);
    }
    grid_barrier(bar_cnt, bar_gen);   // publish h0n to all slices

    // ===== layer 1: h1 = tanh(h0_new W_ih1^T + h1 W_hh1^T + biases + n1) ==
    v8f acc2 = {};
    acc2 = tile_mma(h0n + (size_t)mbase * H_, H_,
                    sWih1 + (size_t)ncol * H_, H_, H_, lane, acc2);
    acc2 = tile_mma(h1c + (size_t)mbase * H_, H_,
                    sWhh1 + (size_t)ncol * H_, H_, H_, lane, acc2);
    for (int r = 0; r < 8; ++r) {
      const int bb = mbase + rowsel + r;
      const float v =
          acc2[r] + bias1 + noise[(((size_t)bb * T_ + t) * 2 + 1) * H_ + ng];
      h1n[(size_t)bb * H_ + ng] = (_Float16)tanhf(v);
    }
    grid_barrier(bar_cnt, bar_gen);   // publish h1n
  }

  // ---- final FC: out = h1 @ fc_w^T + fc_b  ([64,512]x[512,32]) ----
  if (wg == 0) {
    const _Float16* h1f = h1buf + (size_t)(T_ & 1) * (B_ * H_); // buffer 0
    const int nf = ncol + (lane & 15);                          // 0..31
    v8f acc = {};
    acc = tile_mma(h1f + (size_t)mbase * H_, H_,
                   hfc + (size_t)ncol * H_, H_, H_, lane, acc);
    const float bf = fc_b[nf];
    for (int r = 0; r < 8; ++r) {
      const int bb = mbase + rowsel + r;
      out[(size_t)bb * WOUT_ + nf] = acc[r] + bf;
    }
  }
}

// ---------------------------------------------------------------------------
extern "C" void kernel_launch(void* const* d_in, const int* in_sizes, int n_in,
                              void* d_out, int out_size, void* d_ws, size_t ws_size,
                              hipStream_t stream) {
  (void)in_sizes; (void)n_in; (void)out_size; (void)ws_size;
  const float* x     = (const float*)d_in[0];   // [B,T,I]
  const float* noise = (const float*)d_in[1];   // [B,T,2,H]
  const float* W_ih0 = (const float*)d_in[2];   // [H,I]
  const float* W_ih1 = (const float*)d_in[3];   // [H,H]
  const float* W_hh0 = (const float*)d_in[4];   // [H,H]
  const float* W_hh1 = (const float*)d_in[5];   // [H,H]
  const float* b_ih0 = (const float*)d_in[6];
  const float* b_ih1 = (const float*)d_in[7];
  const float* b_hh0 = (const float*)d_in[8];
  const float* b_hh1 = (const float*)d_in[9];
  const float* fc_w  = (const float*)d_in[10];  // [WOUT,H]
  const float* fc_b  = (const float*)d_in[11];

  char* ws = (char*)d_ws;
  _Float16* hWih0 = (_Float16*)(ws + OFF_WIH0);
  _Float16* hWhh0 = (_Float16*)(ws + OFF_WHH0);
  _Float16* hWih1 = (_Float16*)(ws + OFF_WIH1);
  _Float16* hWhh1 = (_Float16*)(ws + OFF_WHH1);
  _Float16* hFcw  = (_Float16*)(ws + OFF_FCW);
  _Float16* hX    = (_Float16*)(ws + OFF_X);
  _Float16* h0buf = (_Float16*)(ws + OFF_H0);
  _Float16* h1buf = (_Float16*)(ws + OFF_H1);
  unsigned* bar   = (unsigned*)(ws + OFF_BAR);

  auto cvt = [&](const float* src, _Float16* dst, int n) {
    int blocks = (n + 1023) / 1024;
    if (blocks > 1024) blocks = 1024;
    cvt_f32_to_f16_kernel<<<blocks, 256, 0, stream>>>(src, dst, n);
  };
  // fp32 -> f16 staging (every call; deterministic)
  cvt(W_ih0, hWih0, H_ * I_);
  cvt(W_hh0, hWhh0, H_ * H_);
  cvt(W_ih1, hWih1, H_ * H_);
  cvt(W_hh1, hWhh1, H_ * H_);
  cvt(fc_w,  hFcw,  WOUT_ * H_);
  cvt(x,     hX,    B_ * T_ * I_);

  // zero hidden-state ping-pong buffers + barrier state (contiguous region)
  {
    int nz = (int)((OFF_BAR + 256 - OFF_H0) / 4);
    zero_u32_kernel<<<(nz + 1023) / 1024, 256, 0, stream>>>(
        (unsigned*)(ws + OFF_H0), nz);
  }

  rnn_persistent<<<NWG, 256, SMEM_BYTES, stream>>>(
      hX, noise, hWih0, hWhh0, hWih1, hWhh1,
      b_ih0, b_ih1, b_hh0, b_hh1,
      hFcw, fc_b, h0buf, h1buf, (float*)d_out, bar, bar + 1);
}
